// TorchCosineIndex_56229711839290
// MI455X (gfx1250) — matrix operand order repaired and checked by reading, not compiled, not run on hardware
//
#include <hip/hip_runtime.h>

// ---------------------------------------------------------------------------
// Cosine top-k on MI455X (gfx1250):
//   pass0 : normalize queries, split into bf16 hi/lo (compensated precision)
//   pass1 : cheap bf16 (hi*hi) WMMA GEMM; per-query tail histogram (values
//           >= T0 only) in LDS -> global; also saves per-row 1/||e||
//   thresh: per-query scan histogram top-down -> threshold (minus margin
//           covering pass1's bf16 error) => provable superset of top-k
//   pass2 : full 3-term split-bf16 WMMA GEMM (hi*hi + hi*lo + lo*hi), tiles
//           staged by the Tensor Data Mover (tensor_load_to_lds, LDS padding
//           via D# pad fields, double buffered, s_wait_tensorcnt); gather
//           (val,idx) >= thr[q] via atomic append
//   pass3 : per-query bitonic sort of candidates in LDS, write top-k
// ---------------------------------------------------------------------------

typedef __attribute__((ext_vector_type(16))) __bf16 v16bf;
typedef __attribute__((ext_vector_type(8)))  __bf16 v8bf;
typedef __attribute__((ext_vector_type(8)))  float  v8f;
typedef __attribute__((ext_vector_type(4)))  unsigned int v4u;
typedef __attribute__((ext_vector_type(8)))  int    v8i;
typedef __attribute__((ext_vector_type(4)))  int    v4i;

#define D       64       // embedding dim
#define NB      1024     // embeddings per block
#define NSUB    128      // embeddings per staged sub-chunk
#define NPAD    4        // LDS row padding (floats) to kill bank conflicts
#define NBINS   64       // histogram bins over [T0, 1]
#define CAP     4096     // candidate capacity per query
#define QMAX    256      // max queries (LDS hist sizing)

#define T0      0.25f                      // histogram floor (tail filter)
#define BINW    ((1.0f - T0) / NBINS)      // bin width
#define MARGIN  0.0125f                    // > 2x pass1 bf16 error bound

static __device__ __forceinline__ v8f wmma_bf16(v16bf a, v16bf b, v8f c) {
    return __builtin_amdgcn_wmma_f32_16x16x32_bf16(
        /*neg_a=*/false, a, /*neg_b=*/false, b,
        /*c_mod=*/(short)0, c, /*reuse_a=*/false, /*reuse_b=*/false);
}

// ---------------- pass0: normalize queries, split bf16 hi/lo ----------------
__global__ void qprep_kernel(const float* __restrict__ q, int B,
                             __bf16* __restrict__ qhi, __bf16* __restrict__ qlo) {
    int r = blockIdx.x * blockDim.x + threadIdx.x;
    if (r >= B) return;
    float ss = 0.f;
    for (int j = 0; j < D; ++j) { float v = q[r * D + j]; ss += v * v; }
    float inv = 1.0f / fmaxf(sqrtf(ss), 1e-12f);
    for (int j = 0; j < D; ++j) {
        float f = q[r * D + j] * inv;
        __bf16 h = (__bf16)f;
        qhi[r * D + j] = h;
        qlo[r * D + j] = (__bf16)(f - (float)h);
    }
}

__global__ void zero_kernel(unsigned int* __restrict__ p, int n) {
    int i = blockIdx.x * blockDim.x + threadIdx.x;
    if (i < n) p[i] = 0u;
}

// ---------------- pass1: hi*hi GEMM + tail histogram + inv-norms ------------
__global__ __launch_bounds__(256)
void hist_pass_kernel(const float* __restrict__ emb,
                      const __bf16* __restrict__ qhi,
                      int N, int B,
                      unsigned int* __restrict__ ghist,
                      float* __restrict__ invn)
{
    __shared__ float sEmb[NSUB][D + NPAD];             // 34 KB
    __shared__ unsigned int sHist[QMAX * NBINS];       // 64 KB

    const int tid   = threadIdx.x;
    const int lane  = tid & 31;
    const int wave  = tid >> 5;
    const int lhalf = lane >> 4;
    const int ln    = lane & 15;

    for (int i = tid; i < QMAX * NBINS; i += 256) sHist[i] = 0u;

    const int blockBase = blockIdx.x * NB;
    const int nqt = B >> 4;

    for (int sub = 0; sub < NB / NSUB; ++sub) {
        const int rowBase = blockBase + sub * NSUB;
        __syncthreads();

        // stage + L2-normalize 128 rows into LDS (2 threads/row); save 1/||e||
        {
            int r = tid >> 1;
            int h = tid & 1;
            int grow = rowBase + r;
            float4 v[8];
            float ss = 0.f;
            if (grow < N) {
                const float4* src = (const float4*)(emb + (size_t)grow * D + h * 32);
#pragma unroll
                for (int j = 0; j < 8; ++j) {
                    v[j] = src[j];
                    ss += v[j].x * v[j].x + v[j].y * v[j].y +
                          v[j].z * v[j].z + v[j].w * v[j].w;
                }
            } else {
#pragma unroll
                for (int j = 0; j < 8; ++j) v[j] = make_float4(0.f, 0.f, 0.f, 0.f);
            }
            ss += __shfl_xor(ss, 1);
            float inv = 1.0f / fmaxf(sqrtf(ss), 1e-12f);
            if (h == 0 && grow < N) invn[grow] = inv;
            float* drow = &sEmb[r][h * 32];
#pragma unroll
            for (int j = 0; j < 8; ++j) {
                drow[4 * j + 0] = v[j].x * inv;
                drow[4 * j + 1] = v[j].y * inv;
                drow[4 * j + 2] = v[j].z * inv;
                drow[4 * j + 3] = v[j].w * inv;
            }
        }
        __syncthreads();

        // B fragments, hi only. B layout: lane<16 K=0..15 (N=lane), lane>=16 K=16..31
        const int ebase = wave * 16;
        v16bf Bhi[2];
#pragma unroll
        for (int kh = 0; kh < 2; ++kh) {
            const float* rowp = &sEmb[ebase + ln][0];
            const float4* p = (const float4*)(rowp + kh * 32 + lhalf * 16);
            float f[16];
#pragma unroll
            for (int c = 0; c < 4; ++c) {
                float4 t = p[c];
                f[4 * c + 0] = t.x; f[4 * c + 1] = t.y;
                f[4 * c + 2] = t.z; f[4 * c + 3] = t.w;
            }
#pragma unroll
            for (int j = 0; j < 16; ++j) Bhi[kh][j] = (__bf16)f[j];
        }

        for (int qt = 0; qt < nqt; ++qt) {
            const int m = qt * 16 + ln;
            v16bf Ahi[2];
#pragma unroll
            for (int kh = 0; kh < 2; ++kh) {
                const int kb = kh * 32 + lhalf * 16;
                v8bf h0 = *(const v8bf*)(qhi + (size_t)m * D + kb);
                v8bf h1 = *(const v8bf*)(qhi + (size_t)m * D + kb + 8);
#pragma unroll
                for (int j = 0; j < 8; ++j) { Ahi[kh][j] = h0[j]; Ahi[kh][8 + j] = h1[j]; }
            }

            v8f acc = {};
            acc = wmma_bf16(Ahi[0], Bhi[0], acc);
            acc = wmma_bf16(Ahi[1], Bhi[1], acc);

#pragma unroll
            for (int r = 0; r < 8; ++r) {
                const int qq = qt * 16 + lhalf * 8 + r;
                const float v = acc[r];
                if (v >= T0) {                       // tail filter: ~2% taken
                    int bin = (int)((v - T0) * (1.0f / BINW));
                    bin = min(bin, NBINS - 1);
                    atomicAdd(&sHist[qq * NBINS + bin], 1u);
                }
            }
        }
    }

    __syncthreads();
    for (int i = tid; i < B * NBINS; i += 256)
        atomicAdd(&ghist[i], sHist[i]);
}

// ---------------- threshold: scan histogram from the top --------------------
__global__ void thresh_kernel(const unsigned int* __restrict__ ghist,
                              const int* __restrict__ kptr, int B,
                              float* __restrict__ thr) {
    int q = blockIdx.x * blockDim.x + threadIdx.x;
    if (q >= B) return;
    const int k = *kptr;
    unsigned int c = 0;
    float t = -1e30f;                 // fallback: gather-all (never hit here)
    for (int b = NBINS - 1; b >= 0; --b) {
        c += ghist[q * NBINS + b];
        if ((int)c >= k) { t = T0 + (float)b * BINW - MARGIN; break; }
    }
    thr[q] = t;
}

// ---------------- pass2: TDM-staged 3-term GEMM + gather --------------------
__global__ __launch_bounds__(256)
void gather_pass_kernel(const float* __restrict__ emb,
                        const __bf16* __restrict__ qhi,
                        const __bf16* __restrict__ qlo,
                        const float* __restrict__ invn,
                        const float* __restrict__ thr,
                        int N, int B,
                        unsigned int* __restrict__ cnt,
                        float* __restrict__ candval,
                        int* __restrict__ candidx)
{
    __shared__ float sBufA[NSUB][D + NPAD];   // 34 KB, TDM double buffer
    __shared__ float sBufB[NSUB][D + NPAD];   // 34 KB

    const int tid   = threadIdx.x;
    const int lane  = tid & 31;
    const int wave  = tid >> 5;
    const int lhalf = lane >> 4;
    const int ln    = lane & 15;

    const int blockBase = blockIdx.x * NB;
    const int nqt = B >> 4;
    const int SUBS = NB / NSUB;

    // Tensor DMA: 128x64 f32 tile, row pad 4 DWORDs every 64 DWORDs so the
    // LDS layout matches sEmb[NSUB][D+NPAD]. OOB rows (>= N) read as zero.
    auto tdm_issue = [&](int rowBase, float* ldsDst) {
        unsigned long long ga = (unsigned long long)(uintptr_t)(emb + (size_t)rowBase * D);
        unsigned int ldsOff = (unsigned int)(uintptr_t)ldsDst;
        int rem = N - rowBase; if (rem < 0) rem = 0;
        v4u g0;
        g0[0] = 1u;                                           // count=1
        g0[1] = ldsOff;                                       // lds_addr
        g0[2] = (unsigned int)(ga & 0xffffffffull);           // global_addr lo
        g0[3] = (unsigned int)((ga >> 32) & 0x01ffffffull)    // global_addr hi
              | (2u << 30);                                   // type=2 (image)
        v8i g1;
        g1[0] = (int)((2u << 16)        // data_size = 4 bytes
                    | (1u << 20)        // pad_enable
                    | (5u << 22)        // pad_interval: every 64 DWORDs
                    | (3u << 25));      // pad_amount: 4 DWORDs
        g1[1] = (int)(64u << 16);                             // tensor_dim0=64
        g1[2] = (int)(((unsigned)rem & 0xffffu) << 16);       // tensor_dim1 lo
        g1[3] = (int)((((unsigned)rem >> 16) & 0xffffu)       // tensor_dim1 hi
                    | (64u << 16));                           // tile_dim0=64
        g1[4] = 128;                                          // tile_dim1=128
        g1[5] = 64;                                           // dim0_stride=64
        g1[6] = 0;
        g1[7] = 0;
        v4i gz = {0, 0, 0, 0};
        v8i gz8 = {0, 0, 0, 0, 0, 0, 0, 0};
        __builtin_amdgcn_tensor_load_to_lds(g0, g1, gz, gz, gz8, 0);
    };

    if (wave == 0) tdm_issue(blockBase, &sBufA[0][0]);

    for (int sub = 0; sub < SUBS; ++sub) {
        const int rowBase = blockBase + sub * NSUB;
        float (*cur)[D + NPAD] = (sub & 1) ? sBufB : sBufA;
        float (*nxt)[D + NPAD] = (sub & 1) ? sBufA : sBufB;

        if (wave == 0) {
            if (sub + 1 < SUBS) {
                tdm_issue(rowBase + NSUB, &nxt[0][0]);        // prefetch next
                __builtin_amdgcn_s_wait_tensorcnt((short)1);  // current done
            } else {
                __builtin_amdgcn_s_wait_tensorcnt((short)0);
            }
        }
        __syncthreads();

        // B fragments, hi/lo, normalization folded in via inv-norm
        const int ebase = wave * 16;
        const int eg = rowBase + ebase + ln;
        const float invE = (eg < N) ? invn[eg] : 0.0f;
        v16bf Bhi[2], Blo[2];
#pragma unroll
        for (int kh = 0; kh < 2; ++kh) {
            const float* rowp = &cur[ebase + ln][0];
            const float4* p = (const float4*)(rowp + kh * 32 + lhalf * 16);
            float f[16];
#pragma unroll
            for (int c = 0; c < 4; ++c) {
                float4 t = p[c];
                f[4 * c + 0] = t.x * invE; f[4 * c + 1] = t.y * invE;
                f[4 * c + 2] = t.z * invE; f[4 * c + 3] = t.w * invE;
            }
#pragma unroll
            for (int j = 0; j < 16; ++j) {
                __bf16 h = (__bf16)f[j];
                Bhi[kh][j] = h;
                Blo[kh][j] = (__bf16)(f[j] - (float)h);
            }
        }

        for (int qt = 0; qt < nqt; ++qt) {
            const int m = qt * 16 + ln;
            v16bf Ahi[2], Alo[2];
#pragma unroll
            for (int kh = 0; kh < 2; ++kh) {
                const int kb = kh * 32 + lhalf * 16;
                v8bf h0 = *(const v8bf*)(qhi + (size_t)m * D + kb);
                v8bf h1 = *(const v8bf*)(qhi + (size_t)m * D + kb + 8);
                v8bf l0 = *(const v8bf*)(qlo + (size_t)m * D + kb);
                v8bf l1 = *(const v8bf*)(qlo + (size_t)m * D + kb + 8);
#pragma unroll
                for (int j = 0; j < 8; ++j) {
                    Ahi[kh][j] = h0[j]; Ahi[kh][8 + j] = h1[j];
                    Alo[kh][j] = l0[j]; Alo[kh][8 + j] = l1[j];
                }
            }

            v8f acc = {};
            acc = wmma_bf16(Ahi[0], Bhi[0], acc);
            acc = wmma_bf16(Ahi[1], Bhi[1], acc);
            acc = wmma_bf16(Ahi[0], Blo[0], acc);
            acc = wmma_bf16(Ahi[1], Blo[1], acc);
            acc = wmma_bf16(Alo[0], Bhi[0], acc);
            acc = wmma_bf16(Alo[1], Bhi[1], acc);

            const int gidx = rowBase + ebase + ln;
#pragma unroll
            for (int r = 0; r < 8; ++r) {
                const int qq = qt * 16 + lhalf * 8 + r;
                const float v = acc[r];
                if (gidx < N && v >= thr[qq]) {
                    unsigned int p = atomicAdd(&cnt[qq], 1u);
                    if (p < CAP) {
                        candval[(size_t)qq * CAP + p] = v;
                        candidx[(size_t)qq * CAP + p] = gidx;
                    }
                }
            }
        }
        __syncthreads();
    }
}

// ---------------- pass3: per-query bitonic sort of candidates ---------------
__global__ __launch_bounds__(256)
void topk_kernel(const float* __restrict__ candval,
                 const int* __restrict__ candidx,
                 const unsigned int* __restrict__ cnt,
                 const int* __restrict__ kptr, int B,
                 float* __restrict__ outbase) {
    __shared__ float sv[CAP];
    __shared__ int   si[CAP];
    const int q = blockIdx.x;
    const int tid = threadIdx.x;
    const int n = min((int)cnt[q], CAP);

    for (int i = tid; i < CAP; i += 256) {
        if (i < n) { sv[i] = candval[(size_t)q * CAP + i]; si[i] = candidx[(size_t)q * CAP + i]; }
        else       { sv[i] = -1e30f; si[i] = -1; }
    }

    for (unsigned int len = 2; len <= CAP; len <<= 1) {
        for (unsigned int s = len >> 1; s > 0; s >>= 1) {
            __syncthreads();
            for (unsigned int i = tid; i < CAP; i += 256) {
                unsigned int j = i ^ s;
                if (j > i) {
                    bool desc = ((i & len) == 0);
                    float a = sv[i], b = sv[j];
                    bool doswap = desc ? (a < b) : (a > b);
                    if (doswap) {
                        sv[i] = b; sv[j] = a;
                        int t = si[i]; si[i] = si[j]; si[j] = t;
                    }
                }
            }
        }
    }
    __syncthreads();

    const int k = *kptr;
    float* outv = outbase;
    int*   outi = ((int*)outbase) + (size_t)B * k;
    for (int i = tid; i < k; i += 256) {
        outv[(size_t)q * k + i] = sv[i];
        outi[(size_t)q * k + i] = si[i];
    }
}

// ---------------------------------------------------------------------------
extern "C" void kernel_launch(void* const* d_in, const int* in_sizes, int n_in,
                              void* d_out, int out_size, void* d_ws, size_t ws_size,
                              hipStream_t stream) {
    const float* dq   = (const float*)d_in[0];
    const float* demb = (const float*)d_in[1];
    const int*   dk   = (const int*)d_in[2];

    const int B = in_sizes[0] / D;   // 256
    const int N = in_sizes[1] / D;   // 1,000,000

    size_t off = 0;
    char* ws = (char*)d_ws;
    auto take = [&](size_t bytes) -> void* {
        void* p = ws + off;
        off += (bytes + 255) & ~(size_t)255;
        return p;
    };
    __bf16* qhi          = (__bf16*)take((size_t)B * D * sizeof(__bf16));
    __bf16* qlo          = (__bf16*)take((size_t)B * D * sizeof(__bf16));
    unsigned int* ghist  = (unsigned int*)take((size_t)B * NBINS * sizeof(unsigned int));
    unsigned int* cnt    = (unsigned int*)take((size_t)B * sizeof(unsigned int));
    float* thr           = (float*)take((size_t)B * sizeof(float));
    float* candval       = (float*)take((size_t)B * CAP * sizeof(float));
    int*   candidx       = (int*)take((size_t)B * CAP * sizeof(int));
    float* invn          = (float*)take((size_t)N * sizeof(float));
    (void)ws_size; (void)n_in; (void)out_size;

    qprep_kernel<<<(B + 255) / 256, 256, 0, stream>>>(dq, B, qhi, qlo);
    zero_kernel<<<(B * NBINS + 255) / 256, 256, 0, stream>>>(ghist, B * NBINS);
    zero_kernel<<<(B + 255) / 256, 256, 0, stream>>>(cnt, B);

    const int nblk = (N + NB - 1) / NB;
    hist_pass_kernel<<<nblk, 256, 0, stream>>>(demb, qhi, N, B, ghist, invn);
    thresh_kernel<<<(B + 255) / 256, 256, 0, stream>>>(ghist, dk, B, thr);
    gather_pass_kernel<<<nblk, 256, 0, stream>>>(demb, qhi, qlo, invn, thr,
                                                 N, B, cnt, candval, candidx);
    topk_kernel<<<B, 256, 0, stream>>>(candval, candidx, cnt, dk, B, (float*)d_out);
}
